// VitSelfAttention_47665547051271
// MI455X (gfx1250) — compile-verified, matrix operand order checked by reading, not run on hardware
//
#include <hip/hip_runtime.h>

// ViT self-attention, fused, bf16 WMMA compute with f32 accumulate.
// Pass 0: convert x and Wq/Wk/Wv to bf16 (Wq pre-scaled by 1/sqrt(H)).
// Pass 1: QKV projection (xb @ Wb^T + b) -> bf16 Q[BH,N,HD], K[BH,N,HD], Vt[BH,HD,N]
//         one wave = 16x64 output tile, 4 accumulators, 24 k-steps of wmma_bf16.
// Pass 2: flash attention per (head, 16-row q tile), online softmax, f32 out.

#define B_  16
#define N_  1024
#define H_  768
#define NH_ 12
#define HD_ 64
#define BH_ (B_ * NH_)               // 192
#define SCALE_ 0.03608439182435161f // 1/sqrt(768)

typedef __attribute__((ext_vector_type(16))) __bf16 v16bf;
typedef __attribute__((ext_vector_type(8)))  __bf16 v8bf;
typedef __attribute__((ext_vector_type(8)))  float  v8f;
typedef __attribute__((ext_vector_type(4)))  float  v4f;

static __device__ __forceinline__ v8f wmma_bf16(v16bf a, v16bf b, v8f c) {
  return __builtin_amdgcn_wmma_f32_16x16x32_bf16(false, a, false, b, (short)0, c,
                                                 false, false);
}

// Load one 16-bf16 WMMA fragment half: elements {0..7} from p, {8..15} from p+16.
// Matches the 16-bit A/B VGPR layout (K pairs packed per VGPR).
static __device__ __forceinline__ v16bf ldfrag(const __bf16* p) {
  return __builtin_shufflevector(*(const v8bf*)p, *(const v8bf*)(p + 16),
                                 0, 1, 2, 3, 4, 5, 6, 7, 8, 9, 10, 11, 12, 13, 14, 15);
}

// ---------------------------------------------------------------------------
// Pass 0: f32 -> bf16 (optionally scaled), 8 elements/thread, vectorized.
// ---------------------------------------------------------------------------
__global__ __launch_bounds__(256) void cvt_bf16_kernel(
    const float* __restrict__ src, __bf16* __restrict__ dst, int n, float scale) {
  const int i = (blockIdx.x * 256 + threadIdx.x) * 8;
  if (i + 8 > n) return;
  v4f a = *(const v4f*)(src + i);
  v4f b = *(const v4f*)(src + i + 4);
  v8bf o;
#pragma unroll
  for (int j = 0; j < 4; ++j) {
    o[j]     = (__bf16)(a[j] * scale);
    o[4 + j] = (__bf16)(b[j] * scale);
  }
  *(v8bf*)(dst + i) = o;
}

// ---------------------------------------------------------------------------
// Pass 1: Y = xb @ Wb^T + bias.  One wave computes a 16x64 output tile
// (4 col tiles sharing the A fragment), K=768 reduction.
// grid = (12 col groups, 128 row groups, 3 matrices), block = 256 (8 waves).
// ---------------------------------------------------------------------------
__global__ __launch_bounds__(256) void qkv_proj_kernel(
    const __bf16* __restrict__ xb, const __bf16* __restrict__ wb,
    const float* __restrict__ bq, const float* __restrict__ bk,
    const float* __restrict__ bv,
    __bf16* __restrict__ qo, __bf16* __restrict__ ko, __bf16* __restrict__ vto) {
  const int lane = threadIdx.x & 31;
  const int wave = threadIdx.x >> 5;
  const int half = lane >> 4;              // 0: K {0-7,16-23}, 1: K {8-15,24-31}
  const int lr   = lane & 15;

  const int colBase = blockIdx.x * 64;            // 0,64,...,704
  const int mTile   = blockIdx.y * 8 + wave;      // 0..1023
  const int mat     = blockIdx.z;                 // 0=Q 1=K 2=V

  const __bf16* W    = wb + (size_t)mat * H_ * H_;
  const float*  bias = (mat == 0) ? bq : (mat == 1) ? bk : bv;
  const float bscale = (mat == 0) ? SCALE_ : 1.0f;  // Wq pre-scaled; scale bq too

  const int mBase = mTile * 16;
  const __bf16* xrow = xb + (size_t)(mBase + lr) * H_ + half * 8;
  const __bf16* w0 = W + (size_t)(colBase + lr) * H_ + half * 8;
  const __bf16* w1 = w0 + (size_t)16 * H_;
  const __bf16* w2 = w0 + (size_t)32 * H_;
  const __bf16* w3 = w0 + (size_t)48 * H_;

  v8f acc0 = {}, acc1 = {}, acc2 = {}, acc3 = {};
  for (int kk = 0; kk < H_; kk += 32) {
    const v16bf a = ldfrag(xrow + kk);
    acc0 = wmma_bf16(a, ldfrag(w0 + kk), acc0);
    acc1 = wmma_bf16(a, ldfrag(w1 + kk), acc1);
    acc2 = wmma_bf16(a, ldfrag(w2 + kk), acc2);
    acc3 = wmma_bf16(a, ldfrag(w3 + kk), acc3);
  }

  const int bidx  = mBase / N_;                   // batch (tile never straddles)
  const int nseq0 = mBase % N_;
  __bf16* dst = (mat == 0) ? qo : (mat == 1) ? ko : vto;

#pragma unroll
  for (int t = 0; t < 4; ++t) {
    const v8f acc = (t == 0) ? acc0 : (t == 1) ? acc1 : (t == 2) ? acc2 : acc3;
    const int nCol = colBase + t * 16 + lr;       // output column for this lane
    const int head = nCol / HD_;
    const int hd   = nCol % HD_;
    const float bcol = bias[nCol] * bscale;
    // Branch-free address: Q/K are [BH,N,HD] (row stride HD), Vt is [BH,HD,N] (stride 1).
    size_t base;
    int    stride;
    if (mat == 2) { base = ((size_t)(bidx * NH_ + head) * HD_ + hd) * N_ + nseq0; stride = 1; }
    else          { base = ((size_t)(bidx * NH_ + head) * N_ + nseq0) * HD_ + hd; stride = HD_; }
#pragma unroll
    for (int j = 0; j < 8; ++j) {
      // C layout: VGPR j, lanes 0-15 -> M=j ; lanes 16-31 -> M=j+8 ; N=lr
      dst[base + (size_t)(j + half * 8) * stride] = (__bf16)(acc[j] + bcol);
    }
  }
}

// ---------------------------------------------------------------------------
// Pass 2: fused flash attention.  One wave owns a 16-row q tile of one head
// and streams 32-key tiles: 4 WMMA for scores, online softmax, P staged
// through LDS (C-layout -> A-layout), 4 WMMA for P·V.
// grid = (BH=192, 16), block = 128 (4 waves, qTile = blockIdx.y*4 + wave).
// ---------------------------------------------------------------------------
__global__ __launch_bounds__(128) void attn_kernel(
    const __bf16* __restrict__ q, const __bf16* __restrict__ k,
    const __bf16* __restrict__ vt, float* __restrict__ out) {
  __shared__ __bf16 plds[4][16][48];  // per-wave 16x32 P tile, 16B-aligned rows

  const int lane = threadIdx.x & 31;
  const int wave = threadIdx.x >> 5;
  const int half = lane >> 4;
  const int lr   = lane & 15;

  const int bh    = blockIdx.x;                    // 0..191
  const int qBase = (blockIdx.y * 4 + wave) * 16;  // 0..1008

  const __bf16* Qb = q  + (size_t)bh * N_ * HD_;
  const __bf16* Kb = k  + (size_t)bh * N_ * HD_;
  const __bf16* Vb = vt + (size_t)bh * HD_ * N_;

  // Q fragments (already scaled by 1/sqrt(H) via Wq/bq): pure b128 loads.
  const __bf16* qrow = Qb + (size_t)(qBase + lr) * HD_ + half * 8;
  const v16bf a0 = ldfrag(qrow);
  const v16bf a1 = ldfrag(qrow + 32);

  float mrun[8], lrun[8];
  v8f o0 = {}, o1 = {}, o2 = {}, o3 = {};
#pragma unroll
  for (int j = 0; j < 8; ++j) { mrun[j] = -3.0e38f; lrun[j] = 0.0f; }

  for (int kb = 0; kb < N_; kb += 32) {
    if (kb + 32 < N_)
      __builtin_prefetch(Kb + (size_t)(kb + 32 + lr) * HD_, 0, 1);

    // K fragments: keys kb+lr (low tile) and kb+16+lr (high tile), contiguous rows.
    const __bf16* klo = Kb + (size_t)(kb + lr) * HD_ + half * 8;
    const __bf16* khi = klo + (size_t)16 * HD_;

    v8f z = {};
    v8f slo = wmma_bf16(a0, ldfrag(klo), z);
    slo     = wmma_bf16(a1, ldfrag(klo + 32), slo);
    v8f shi = wmma_bf16(a0, ldfrag(khi), z);
    shi     = wmma_bf16(a1, ldfrag(khi + 32), shi);

    // Online softmax over 32 columns; rows live per-VGPR j within 16-lane halves.
#pragma unroll
    for (int j = 0; j < 8; ++j) {
      float mt = fmaxf(slo[j], shi[j]);
#pragma unroll
      for (int off = 8; off >= 1; off >>= 1) mt = fmaxf(mt, __shfl_xor(mt, off, 16));
      const float mnew  = fmaxf(mrun[j], mt);
      const float alpha = __expf(mrun[j] - mnew);
      const float pl = __expf(slo[j] - mnew);
      const float ph = __expf(shi[j] - mnew);
      slo[j] = pl; shi[j] = ph;
      float rs = pl + ph;
#pragma unroll
      for (int off = 8; off >= 1; off >>= 1) rs += __shfl_xor(rs, off, 16);
      lrun[j] = lrun[j] * alpha + rs;
      mrun[j] = mnew;
      o0[j] *= alpha; o1[j] *= alpha; o2[j] *= alpha; o3[j] *= alpha;
    }

    // Stage P (16x32, C-layout) into LDS, reread in A-layout.
    // Same-wave DS ops are architecturally in-order; just block compiler reorder.
#pragma unroll
    for (int j = 0; j < 8; ++j) {
      const int row = j + half * 8;
      plds[wave][row][lr]      = (__bf16)slo[j];
      plds[wave][row][16 + lr] = (__bf16)shi[j];
    }
    asm volatile("" ::: "memory");
    const v16bf pa = ldfrag(&plds[wave][lr][half * 8]);
    asm volatile("" ::: "memory");

    // P·V: B[k][d] columns are rows of Vt (contiguous along keys).
    const __bf16* vrow = Vb + (size_t)lr * N_ + kb + half * 8;
    o0 = wmma_bf16(pa, ldfrag(vrow), o0);
    o1 = wmma_bf16(pa, ldfrag(vrow + (size_t)16 * N_), o1);
    o2 = wmma_bf16(pa, ldfrag(vrow + (size_t)32 * N_), o2);
    o3 = wmma_bf16(pa, ldfrag(vrow + (size_t)48 * N_), o3);
  }

  // Normalize and store ctx back as [B, N, NH*HD] f32.
  const int bidx = bh / NH_, head = bh % NH_;
  float* outb = out + (size_t)bidx * N_ * H_ + (size_t)head * HD_;
#pragma unroll
  for (int j = 0; j < 8; ++j) {
    const float inv = 1.0f / lrun[j];
    float* orow = outb + (size_t)(qBase + j + half * 8) * H_;
    orow[lr]      = o0[j] * inv;
    orow[16 + lr] = o1[j] * inv;
    orow[32 + lr] = o2[j] * inv;
    orow[48 + lr] = o3[j] * inv;
  }
}

// ---------------------------------------------------------------------------
extern "C" void kernel_launch(void* const* d_in, const int* in_sizes, int n_in,
                              void* d_out, int out_size, void* d_ws, size_t ws_size,
                              hipStream_t stream) {
  const float* x  = (const float*)d_in[0];
  const float* Wq = (const float*)d_in[1];
  const float* bq = (const float*)d_in[2];
  const float* Wk = (const float*)d_in[3];
  const float* bk = (const float*)d_in[4];
  const float* Wv = (const float*)d_in[5];
  const float* bv = (const float*)d_in[6];
  float* out = (float*)d_out;

  const size_t qkvElems = (size_t)BH_ * N_ * HD_;  // 12,582,912 (25 MB bf16 each)
  const size_t xElems   = (size_t)B_ * N_ * H_;    // 12,582,912
  const size_t wElems   = (size_t)H_ * H_;         // 589,824

  __bf16* qbuf  = (__bf16*)d_ws;
  __bf16* kbuf  = qbuf + qkvElems;
  __bf16* vtbuf = kbuf + qkvElems;
  __bf16* xbuf  = vtbuf + qkvElems;
  __bf16* wbuf  = xbuf + xElems;                   // 3 matrices, contiguous
  // total d_ws: (3*qkv + x + 3*W) * 2B  ~= 104 MB

  // Pass 0: conversions (Wq pre-scaled by 1/sqrt(H)).
  cvt_bf16_kernel<<<dim3((unsigned)(xElems / (8 * 256))), 256, 0, stream>>>(
      x, xbuf, (int)xElems, 1.0f);
  cvt_bf16_kernel<<<dim3((unsigned)(wElems / (8 * 256))), 256, 0, stream>>>(
      Wq, wbuf, (int)wElems, SCALE_);
  cvt_bf16_kernel<<<dim3((unsigned)(wElems / (8 * 256))), 256, 0, stream>>>(
      Wk, wbuf + wElems, (int)wElems, 1.0f);
  cvt_bf16_kernel<<<dim3((unsigned)(wElems / (8 * 256))), 256, 0, stream>>>(
      Wv, wbuf + 2 * wElems, (int)wElems, 1.0f);

  // Pass 1: QKV projection.
  dim3 g1(H_ / 64, (B_ * N_) / (16 * 8), 3);       // (12, 128, 3)
  qkv_proj_kernel<<<g1, 256, 0, stream>>>(xbuf, wbuf, bq, bk, bv,
                                          qbuf, kbuf, vtbuf);

  // Pass 2: fused attention.
  dim3 g2(BH_, N_ / (16 * 4));                     // (192, 16)
  attn_kernel<<<g2, 128, 0, stream>>>(qbuf, kbuf, vtbuf, out);
}
